// CausalSelfAttention_47854525612243
// MI455X (gfx1250) — matrix operand order repaired and checked by reading, not compiled, not run on hardware
//
#include <hip/hip_runtime.h>
#include <hip/hip_bf16.h>

// ---------------------------------------------------------------------------
// CDNA5 (gfx1250) GQA causal attention with sliding window + ALiBi.
// bf16 WMMA (v_wmma_f32_16x16x32_bf16) everywhere, f32 accumulate.
// Attention K/V tiles staged to LDS via async global->LDS copies (ASYNCcnt),
// double-buffered.
// ---------------------------------------------------------------------------

typedef __bf16 bf16;
typedef __attribute__((ext_vector_type(16))) __bf16 v16bf;
typedef __attribute__((ext_vector_type(8)))  float  v8f;
typedef int v4i __attribute__((vector_size(4 * sizeof(int))));

#define S_LEN   2048
#define D_MODEL 2048
#define N_HEADS 16
#define N_KV    4
#define HD      128
#define WINDOW  512
#define KV_COLS (N_KV * HD)     // 512

// ---- workspace layout (bytes) ----
static constexpr size_t OFF_XB  = 0;
static constexpr size_t OFF_WQT = OFF_XB  + (size_t)D_MODEL * D_MODEL * 2;
static constexpr size_t OFF_WOT = OFF_WQT + (size_t)D_MODEL * D_MODEL * 2;
static constexpr size_t OFF_WKT = OFF_WOT + (size_t)D_MODEL * D_MODEL * 2;
static constexpr size_t OFF_WVT = OFF_WKT + (size_t)KV_COLS * D_MODEL * 2;
static constexpr size_t OFF_QB  = OFF_WVT + (size_t)KV_COLS * D_MODEL * 2;
static constexpr size_t OFF_KB  = OFF_QB  + (size_t)S_LEN * D_MODEL * 2;
static constexpr size_t OFF_VT  = OFF_KB  + (size_t)S_LEN * KV_COLS * 2;
static constexpr size_t OFF_YB  = OFF_VT  + (size_t)KV_COLS * S_LEN * 2;

// ---------------------------------------------------------------------------
// Async global->LDS staging (gfx1250 ASYNCcnt path), with sync fallback.
// Builtin signature (from hipcc diagnostic): (v4i AS1*, v4i AS3*, imm, imm).
// ---------------------------------------------------------------------------
#ifndef __has_builtin
#define __has_builtin(x) 0
#endif

#define TO_GLB(p) ((__attribute__((address_space(1))) v4i*)(p))
#define TO_LDS(p) ((__attribute__((address_space(3))) v4i*)(p))

#if __has_builtin(__builtin_amdgcn_global_load_async_to_lds_b128)
#define HAVE_ASYNC_LDS 1
#define STAGE_CP(g, l) \
    __builtin_amdgcn_global_load_async_to_lds_b128(TO_GLB(g), TO_LDS(l), 0, 0)
#if __has_builtin(__builtin_amdgcn_s_wait_asynccnt)
#define WAIT_ASYNC(n) __builtin_amdgcn_s_wait_asynccnt(n)
#else
#define WAIT_ASYNC(n) asm volatile("s_wait_asynccnt %0" ::"i"(n) : "memory")
#endif
#else
#define HAVE_ASYNC_LDS 0
#define STAGE_CP(g, l) (*(uint4*)(l) = *(const uint4*)(g))
#define WAIT_ASYNC(n) ((void)0)
#endif

// ---------------------------------------------------------------------------
// Fragment helpers (contiguous k order chosen here, same for A and B):
// A (16x32 bf16):  lane {hi=lane>>4, m=lane&15}; halves j -> K = kb+j,
//                  kb incl hi*16.
// B (32x16 bf16):  lane {hi, n=lane&15}; halves j -> K = kb+j, kb incl hi*16.
// C/D (16x16 f32): VGPR r, lane -> M = r + 8*(lane>>4), N = lane&15.
// ---------------------------------------------------------------------------
__device__ __forceinline__ v16bf ld_frag_split(const bf16* p) {
    union { uint4 q[2]; v16bf v; } u;
    u.q[0] = *(const uint4*)(p);
    u.q[1] = *(const uint4*)(p + 8);
    return u.v;
}
__device__ __forceinline__ v16bf ld_frag_contig(const bf16* p) {
    union { uint4 q[2]; v16bf v; } u;
    u.q[0] = *(const uint4*)(p);
    u.q[1] = *(const uint4*)(p + 8);
    return u.v;
}
__device__ __forceinline__ v8f wmma_bf16(v16bf a, v16bf b, v8f c) {
    return __builtin_amdgcn_wmma_f32_16x16x32_bf16(
        false, a, false, b, (short)0, c, false, false);
}

// ---------------------------------------------------------------------------
// Conversion kernels
// ---------------------------------------------------------------------------
__global__ void cvt_f32_bf16_kernel(const float* __restrict__ src,
                                    bf16* __restrict__ dst, int n) {
    int i = blockIdx.x * blockDim.x + threadIdx.x;
    if (i < n) dst[i] = (bf16)src[i];
}

// src[K][N] f32  ->  dst[N][K] bf16
__global__ void transpose_cvt_kernel(const float* __restrict__ src,
                                     bf16* __restrict__ dst, int K, int N) {
    int i = blockIdx.x * blockDim.x + threadIdx.x;
    if (i < K * N) {
        int n = i / K;
        int k = i - n * K;
        dst[i] = (bf16)src[k * N + n];
    }
}

// ---------------------------------------------------------------------------
// GEMM: C[M][N] = A[M][K] @ B[K][N], B given transposed (Bt[N][K]).
// 256 threads = 8 waves; wave w computes rows [m0, m0+32) x cols [n0, n0+128):
// two A fragments share each B fragment (16 WMMA / k-step).
// MODE 0: bf16 row-major (acc*scale); MODE 1: bf16 transposed out[N][M];
// MODE 2: f32 row-major.
// ---------------------------------------------------------------------------
template <int MODE>
__global__ __launch_bounds__(256)
void gemm_bf16_kernel(const bf16* __restrict__ A, const bf16* __restrict__ Bt,
                      void* __restrict__ outp, int M, int N, int K, float scale) {
    const int lane = threadIdx.x & 31;
    const int w    = threadIdx.x >> 5;
    const int nl   = lane & 15;
    const int hi   = lane >> 4;
    const int m0   = blockIdx.x * 256 + w * 32;
    const int n0   = blockIdx.y * 128;

    v8f acc0[8], acc1[8];
#pragma unroll
    for (int c = 0; c < 8; ++c) { acc0[c] = (v8f)0.0f; acc1[c] = (v8f)0.0f; }

    const bf16* arow0 = A + (size_t)(m0 + nl) * K + hi * 16;
    const bf16* arow1 = arow0 + (size_t)16 * K;
    for (int kc = 0; kc < K; kc += 32) {
        v16bf a0 = ld_frag_split(arow0 + kc);
        v16bf a1 = ld_frag_split(arow1 + kc);
#pragma unroll
        for (int c = 0; c < 8; ++c) {
            v16bf b = ld_frag_contig(Bt + (size_t)(n0 + c * 16 + nl) * K + kc + hi * 16);
            acc0[c] = wmma_bf16(a0, b, acc0[c]);
            acc1[c] = wmma_bf16(a1, b, acc1[c]);
        }
    }

#pragma unroll
    for (int t = 0; t < 2; ++t) {
#pragma unroll
        for (int c = 0; c < 8; ++c) {
#pragma unroll
            for (int r = 0; r < 8; ++r) {
                float v = (t == 0) ? acc0[c][r] : acc1[c][r];
                int row = m0 + t * 16 + r + 8 * hi;
                int col = n0 + c * 16 + nl;
                if (MODE == 0) {
                    ((bf16*)outp)[(size_t)row * N + col] = (bf16)(v * scale);
                } else if (MODE == 1) {
                    ((bf16*)outp)[(size_t)col * M + row] = (bf16)v;
                } else {
                    ((float*)outp)[(size_t)row * N + col] = v;
                }
            }
        }
    }
}

// ---------------------------------------------------------------------------
// Flash attention: one wave per (head, 16-row q tile). K/V tiles staged to
// LDS with async copies, double buffered.
// ---------------------------------------------------------------------------
#define KROW 136   // 128 + 8 bf16 pad (272 B rows -> bank rotation)
#define VROW 40    // 32 + 8 bf16 pad  (80 B rows)

__device__ __forceinline__ void stage_tiles(const bf16* __restrict__ kg,  // Kb + kv*HD
                                            const bf16* __restrict__ vg,  // Vt + kv*HD*S
                                            int kb, int lane,
                                            bf16 (*Klb)[KROW], bf16 (*Vlb)[VROW]) {
    // K tile: 32 keys x 128 bf16 (row = key), 16 b128 transfers x 32 lanes
#pragma unroll
    for (int i = 0; i < 16; ++i) {
        int t   = i * 32 + lane;
        int key = t >> 4;
        int ch  = t & 15;
        STAGE_CP(kg + (size_t)(kb + key) * KV_COLS + ch * 8, &Klb[key][ch * 8]);
    }
    // V tile: 128 d-rows x 32 keys (row = d), 16 b128 transfers x 32 lanes
#pragma unroll
    for (int i = 0; i < 16; ++i) {
        int t  = i * 32 + lane;
        int d  = t >> 2;
        int ch = t & 3;
        STAGE_CP(vg + (size_t)d * S_LEN + kb + ch * 8, &Vlb[d][ch * 8]);
    }
}

__global__ __launch_bounds__(32)
void attn_kernel(const bf16* __restrict__ Qb,   // [S][NH*HD]  (pre-scaled)
                 const bf16* __restrict__ Kb,   // [S][NKV*HD]
                 const bf16* __restrict__ Vt,   // [NKV*HD][S]
                 bf16* __restrict__ Yb) {       // [S][NH*HD]
    const int lane = threadIdx.x & 31;
    const int nl   = lane & 15;
    const int hi   = lane >> 4;
    const int rg   = hi * 8;
    const int h    = blockIdx.y;
    const int kvh  = h >> 2;
    const int q0   = blockIdx.x * 16;
    const float slope = exp2f(-0.5f * (float)(h + 1));  // ALiBi, n_heads=16

    __shared__ bf16 Kl[2][32][KROW];
    __shared__ bf16 Vl[2][128][VROW];
    __shared__ bf16 P[16][32];

    const bf16* kg = Kb + kvh * HD;
    const bf16* vg = Vt + (size_t)(kvh * HD) * S_LEN;

    // 16x128 Q tile as 4 A-fragments
    v16bf qa[4];
#pragma unroll
    for (int c = 0; c < 4; ++c)
        qa[c] = ld_frag_split(Qb + (size_t)(q0 + nl) * D_MODEL + h * HD + c * 32 + hi * 16);

    v8f O[8];
#pragma unroll
    for (int c = 0; c < 8; ++c) O[c] = (v8f)0.0f;
    float mrow[8], lrow[8];
#pragma unroll
    for (int r = 0; r < 8; ++r) { mrow[r] = -1e30f; lrow[r] = 0.0f; }

    int kstart = q0 - (WINDOW - 1);
    if (kstart < 0) kstart = 0;
    kstart &= ~31;

    // prologue: stage first tile into buffer 0
    stage_tiles(kg, vg, kstart, lane, Kl[0], Vl[0]);

    int buf = 0;
    for (int kb = kstart; kb <= q0 + 15; kb += 32) {
        const bool pf = (kb + 32 <= q0 + 15);
        if (pf) stage_tiles(kg, vg, kb + 32, lane, Kl[buf ^ 1], Vl[buf ^ 1]);
        if (pf) { WAIT_ASYNC(32); } else { WAIT_ASYNC(0); }
        __syncthreads();

        bf16 (*Klb)[KROW] = Kl[buf];
        bf16 (*Vlb)[VROW] = Vl[buf];

        // ---- scores: two 16x16 tiles (keys kb..kb+15, kb+16..kb+31) ----
        v8f s0 = (v8f)0.0f, s1 = (v8f)0.0f;
#pragma unroll
        for (int c = 0; c < 4; ++c) {
            v16bf b0 = ld_frag_contig(&Klb[nl][c * 32 + hi * 16]);
            v16bf b1 = ld_frag_contig(&Klb[nl + 16][c * 32 + hi * 16]);
            s0 = wmma_bf16(qa[c], b0, s0);
            s1 = wmma_bf16(qa[c], b1, s1);
        }

        // ---- mask + ALiBi + online softmax ----
#pragma unroll
        for (int r = 0; r < 8; ++r) {
            const int q  = q0 + r + rg;
            const int k0 = kb + nl;
            const int k1 = k0 + 16;
            const bool ok0 = (k0 <= q) && (k0 > q - WINDOW);
            const bool ok1 = (k1 <= q) && (k1 > q - WINDOW);
            float f0 = ok0 ? (s0[r] - slope * (float)(q - k0)) : -1e30f;
            float f1 = ok1 ? (s1[r] - slope * (float)(q - k1)) : -1e30f;

            float tm = fmaxf(f0, f1);
#pragma unroll
            for (int off = 1; off < 16; off <<= 1)
                tm = fmaxf(tm, __shfl_xor(tm, off, 16));

            const float nm   = fmaxf(mrow[r], tm);
            const float corr = __expf(mrow[r] - nm);
            mrow[r] = nm;
            const float p0 = ok0 ? __expf(f0 - nm) : 0.0f;
            const float p1 = ok1 ? __expf(f1 - nm) : 0.0f;
            float rs = p0 + p1;
#pragma unroll
            for (int off = 1; off < 16; off <<= 1)
                rs += __shfl_xor(rs, off, 16);
            lrow[r] = lrow[r] * corr + rs;
#pragma unroll
            for (int c = 0; c < 8; ++c) O[c][r] *= corr;

            P[r + rg][nl]      = (bf16)p0;
            P[r + rg][nl + 16] = (bf16)p1;
        }
        __syncthreads();

        // ---- PV: O(16x128) += P(16x32) @ V(32x128) ----
        v16bf pa = ld_frag_split(&P[nl][hi * 16]);
#pragma unroll
        for (int c = 0; c < 8; ++c) {
            v16bf vb = ld_frag_contig(&Vlb[c * 16 + nl][hi * 16]);
            O[c] = wmma_bf16(pa, vb, O[c]);
        }
        __syncthreads();
        buf ^= 1;
    }

    // ---- normalize + store Y ----
    float inv[8];
#pragma unroll
    for (int r = 0; r < 8; ++r) inv[r] = 1.0f / lrow[r];
#pragma unroll
    for (int c = 0; c < 8; ++c) {
#pragma unroll
        for (int r = 0; r < 8; ++r) {
            Yb[(size_t)(q0 + r + rg) * D_MODEL + h * HD + c * 16 + nl] =
                (bf16)(O[c][r] * inv[r]);
        }
    }
}

// ---------------------------------------------------------------------------
// Host-side launch sequence
// ---------------------------------------------------------------------------
extern "C" void kernel_launch(void* const* d_in, const int* in_sizes, int n_in,
                              void* d_out, int out_size, void* d_ws, size_t ws_size,
                              hipStream_t stream) {
    (void)in_sizes; (void)n_in; (void)out_size; (void)ws_size;
    const float* x  = (const float*)d_in[0];
    const float* Wq = (const float*)d_in[1];
    const float* Wk = (const float*)d_in[2];
    const float* Wv = (const float*)d_in[3];
    const float* Wo = (const float*)d_in[4];

    char* ws = (char*)d_ws;
    bf16* xb  = (bf16*)(ws + OFF_XB);
    bf16* wqt = (bf16*)(ws + OFF_WQT);
    bf16* wot = (bf16*)(ws + OFF_WOT);
    bf16* wkt = (bf16*)(ws + OFF_WKT);
    bf16* wvt = (bf16*)(ws + OFF_WVT);
    bf16* qb  = (bf16*)(ws + OFF_QB);
    bf16* kb  = (bf16*)(ws + OFF_KB);
    bf16* vt  = (bf16*)(ws + OFF_VT);
    bf16* yb  = (bf16*)(ws + OFF_YB);

    const int nX = S_LEN * D_MODEL;
    cvt_f32_bf16_kernel<<<(nX + 255) / 256, 256, 0, stream>>>(x, xb, nX);

    const int nQW = D_MODEL * D_MODEL;
    const int nKW = D_MODEL * KV_COLS;
    transpose_cvt_kernel<<<(nQW + 255) / 256, 256, 0, stream>>>(Wq, wqt, D_MODEL, D_MODEL);
    transpose_cvt_kernel<<<(nKW + 255) / 256, 256, 0, stream>>>(Wk, wkt, D_MODEL, KV_COLS);
    transpose_cvt_kernel<<<(nKW + 255) / 256, 256, 0, stream>>>(Wv, wvt, D_MODEL, KV_COLS);
    transpose_cvt_kernel<<<(nQW + 255) / 256, 256, 0, stream>>>(Wo, wot, D_MODEL, D_MODEL);

    const float qscale = 0.088388347648318447f;  // 1/sqrt(128)

    // Q = x @ Wq (pre-scaled), bf16 row-major [S][2048]
    gemm_bf16_kernel<0><<<dim3(S_LEN / 256, D_MODEL / 128), 256, 0, stream>>>(
        xb, wqt, qb, S_LEN, D_MODEL, D_MODEL, qscale);
    // K = x @ Wk, bf16 row-major [S][512]
    gemm_bf16_kernel<0><<<dim3(S_LEN / 256, KV_COLS / 128), 256, 0, stream>>>(
        xb, wkt, kb, S_LEN, KV_COLS, D_MODEL, 1.0f);
    // V = x @ Wv, stored transposed: Vt[512][S]
    gemm_bf16_kernel<1><<<dim3(S_LEN / 256, KV_COLS / 128), 256, 0, stream>>>(
        xb, wvt, vt, S_LEN, KV_COLS, D_MODEL, 1.0f);

    // attention: grid = (q tiles, heads), 1 wave per block
    attn_kernel<<<dim3(S_LEN / 16, N_HEADS), 32, 0, stream>>>(qb, kb, vt, yb);

    // out = Y @ Wo, f32 row-major to d_out
    gemm_bf16_kernel<2><<<dim3(S_LEN / 256, D_MODEL / 128), 256, 0, stream>>>(
        yb, wot, d_out, S_LEN, D_MODEL, D_MODEL, 1.0f);
}
